// GAT_41326175322935
// MI455X (gfx1250) — matrix operand order repaired and checked by reading, not compiled, not run on hardware
//
#include <hip/hip_runtime.h>
#include <hip/hip_bf16.h>
#include <stdint.h>

// ---------------- problem constants (from reference) ----------------
#define NN_NODES 100000
#define NE_EDGES 1600000LL
#define ET_EDGES (NE_EDGES + (long long)NN_NODES)   // + self loops
#define F_IN     128
#define HID1     128                                // HEADS*HID = 4*32
#define HEADS1   4
#define C1       32
#define OUT2DIM  64

typedef __attribute__((ext_vector_type(16))) _Float16 v16h;
typedef __attribute__((ext_vector_type(8)))  float    v8f;

// K-swizzle within each 32-wide group: [0..7, 16..23, 8..15, 24..31].
// Lets a lane's 16-element A fragment be one contiguous 32B load.
__device__ __forceinline__ int kswz(int k) {
    int g = k & ~31, j = k & 31;
    int nj = (j < 8) ? j : (j < 16 ? j + 8 : (j < 24 ? j - 8 : j));
    return g + nj;
}

// ---------------- small utility kernels ----------------
__global__ void fill_f32_k(float* __restrict__ p, float v, long long n) {
    long long i = (long long)blockIdx.x * blockDim.x + threadIdx.x;
    if (i < n) p[i] = v;
}

// fp32 -> f16 with per-row K-swizzle (rows of 128)
__global__ void cvt_f16_swz_k(const float* __restrict__ s, _Float16* __restrict__ d, long long n) {
    long long i = (long long)blockIdx.x * blockDim.x + threadIdx.x;
    if (i >= n) return;
    long long base = i & ~(long long)127;
    int k = (int)(i & 127);
    d[base + kswz(k)] = (_Float16)s[i];
}

// plain fp32 -> f16 (for weights; GEMM swizzles them during LDS fill)
__global__ void cvt_f16_k(const float* __restrict__ s, _Float16* __restrict__ d, long long n) {
    long long i = (long long)blockIdx.x * blockDim.x + threadIdx.x;
    if (i < n) d[i] = (_Float16)s[i];
}

__device__ __forceinline__ void atomicMaxF(float* addr, float val) {
    // IEEE-monotone trick: emits global_atomic_max_i32 / global_atomic_min_u32
    if (val >= 0.f) atomicMax((int*)addr, __float_as_int(val));
    else            atomicMin((unsigned int*)addr, __float_as_uint(val));
}

// ---------------- WMMA GEMM: Dst[NN,NOUT](f32) = A[NN,128](f16,K-swizzled) @ W[128,NOUT](f16) ----
template<int NOUT>
__launch_bounds__(256)
__global__ void gat_gemm_wmma_k(const _Float16* __restrict__ A,
                                const _Float16* __restrict__ W,
                                float* __restrict__ Dst,
                                int NN)
{
    constexpr int NT = NOUT / 16;
    // W stored in LDS as ready-made B fragments:
    // frag f = (kk/32)*NT + t; within frag: lane*16 + elem (32B contiguous per lane)
    __shared__ alignas(32) _Float16 wlds[128 * NOUT];
    for (int q = threadIdx.x; q < 128 * NOUT; q += 256) {
        int f = q >> 9;            // /512 halves per frag
        int r = q & 511;
        int l = r >> 4;            // lane
        int i = r & 15;            // element
        int kk = (f / NT) << 5;
        int t  = f - (f / NT) * NT;
        int k  = kk + ((i >> 3) << 4) + ((l >> 4) << 3) + (i & 7);   // lanes 0-15: K kk+{0..7,16..23}; 16-31: kk+{8..15,24..31}
        int c  = (t << 4) + (l & 15);
        wlds[q] = W[k * NOUT + c];
    }
    __syncthreads();

    const int lane = threadIdx.x & 31;
    const int wave = threadIdx.x >> 5;
    const int m0   = blockIdx.x * 128 + wave * 16; // this wave's 16-row strip
    const int kh   = lane >> 4;                    // K-half selector
    const int lc   = lane & 15;

    int r = m0 + lc;
    if (r >= NN) r = NN - 1;                       // clamp loads; stores masked below
    const _Float16* arow = A + (size_t)r * 128;

    v8f acc[NT];
#pragma unroll
    for (int t = 0; t < NT; t++) acc[t] = (v8f)0.f;

#pragma unroll
    for (int kk = 0; kk < 128; kk += 32) {
        // A fragment: one aligned 32B load thanks to the K-swizzled activation layout
        const v16h a = *(const v16h*)(arow + kk + 16 * kh);
#pragma unroll
        for (int t = 0; t < NT; t++) {
            const v16h b = *(const v16h*)(&wlds[(((kk >> 5) * NT + t) << 9) + (lane << 4)]);
            acc[t] = __builtin_amdgcn_wmma_f32_16x16x32_f16(
                false, a, false, b, (short)0, acc[t], false, false);
        }
    }

    // C/D layout: lane covers column lc of rows m0+{0..7} (lanes<16) or m0+{8..15}
    const int rbase = m0 + (kh ? 8 : 0);
#pragma unroll
    for (int t = 0; t < NT; t++) {
        const int c = t * 16 + lc;
#pragma unroll
        for (int v = 0; v < 8; v++) {
            const int row = rbase + v;
            if (row < NN) Dst[(size_t)row * NOUT + c] = acc[t][v];
        }
    }
}

// ---------------- per-node attention dots ----------------
__global__ void nalpha1_k(const float* __restrict__ h, const float* __restrict__ asrc,
                          const float* __restrict__ adst, float* __restrict__ es,
                          float* __restrict__ ed, int N)
{
    int t = blockIdx.x * blockDim.x + threadIdx.x;
    if (t >= N * HEADS1) return;
    int n = t >> 2, hd = t & 3;
    const float* hp = h + (size_t)n * HID1 + hd * C1;
    const float* as = asrc + hd * C1;
    const float* ad = adst + hd * C1;
    float s1 = 0.f, s2 = 0.f;
#pragma unroll
    for (int c = 0; c < C1; c++) { float v = hp[c]; s1 += v * as[c]; s2 += v * ad[c]; }
    es[t] = s1; ed[t] = s2;
}

__global__ void nalpha2_k(const float* __restrict__ g, const float* __restrict__ asrc,
                          const float* __restrict__ adst, float* __restrict__ es,
                          float* __restrict__ ed, int N)
{
    int n = blockIdx.x * blockDim.x + threadIdx.x;
    if (n >= N) return;
    const float* gp = g + (size_t)n * OUT2DIM;
    float s1 = 0.f, s2 = 0.f;
#pragma unroll
    for (int c = 0; c < OUT2DIM; c++) { float v = gp[c]; s1 += v * asrc[c]; s2 += v * adst[c]; }
    es[n] = s1; ed[n] = s2;
}

// ---------------- per-edge passes ----------------
template<int HH>
__global__ void edge_max_k(const long long* __restrict__ ei, long long E, long long ET,
                           const float* __restrict__ es, const float* __restrict__ ed,
                           float* __restrict__ alpha, float* __restrict__ amax)
{
    long long e = (long long)blockIdx.x * blockDim.x + threadIdx.x;
    if (e >= ET) return;
    long long s, d;
    if (e < E) { s = ei[e]; d = ei[E + e]; } else { s = d = e - E; }
#pragma unroll
    for (int h = 0; h < HH; h++) {
        float a = es[s * HH + h] + ed[d * HH + h];
        a = a > 0.f ? a : 0.2f * a;                 // leaky_relu(0.2)
        alpha[e * HH + h] = a;
        atomicMaxF(&amax[d * HH + h], a);
    }
}

template<int HH>
__global__ void edge_exp_k(const long long* __restrict__ ei, long long E, long long ET,
                           float* __restrict__ alpha, const float* __restrict__ amax,
                           float* __restrict__ den)
{
    long long e = (long long)blockIdx.x * blockDim.x + threadIdx.x;
    if (e >= ET) return;
    long long d = (e < E) ? ei[E + e] : (e - E);
#pragma unroll
    for (int h = 0; h < HH; h++) {
        float ex = __expf(alpha[e * HH + h] - amax[d * HH + h]);
        alpha[e * HH + h] = ex;
        atomicAdd(&den[d * HH + h], ex);
    }
}

// one wave per edge; lane owns 4 channels (layer1: 128 ch, 4 heads)
__global__ void agg1_k(const long long* __restrict__ ei, long long E, long long ET,
                       const float* __restrict__ h1, const float* __restrict__ ex,
                       const float* __restrict__ den, float* __restrict__ out)
{
    const int lane = threadIdx.x & 31;
    long long e = (long long)blockIdx.x * 8 + (threadIdx.x >> 5);
    if (e >= ET) return;
    long long s, d;
    if (e < E) { s = ei[e]; d = ei[E + e]; } else { s = d = e - E; }
    const int c0 = lane * 4;
    const int hd = c0 >> 5;
    const float coef = ex[e * 4 + hd] / den[d * 4 + hd];
    const float4 v = *(const float4*)(h1 + (size_t)s * HID1 + c0);
    float* o = out + (size_t)d * HID1 + c0;
    atomicAdd(o + 0, v.x * coef);
    atomicAdd(o + 1, v.y * coef);
    atomicAdd(o + 2, v.z * coef);
    atomicAdd(o + 3, v.w * coef);
}

// one wave per edge; lane owns 2 channels (layer2: 64 ch, 1 head)
__global__ void agg2_k(const long long* __restrict__ ei, long long E, long long ET,
                       const float* __restrict__ g2, const float* __restrict__ ex,
                       const float* __restrict__ den, float* __restrict__ out)
{
    const int lane = threadIdx.x & 31;
    long long e = (long long)blockIdx.x * 8 + (threadIdx.x >> 5);
    if (e >= ET) return;
    long long s, d;
    if (e < E) { s = ei[e]; d = ei[E + e]; } else { s = d = e - E; }
    const int c0 = lane * 2;
    const float coef = ex[e] / den[d];
    const float2 v = *(const float2*)(g2 + (size_t)s * OUT2DIM + c0);
    float* o = out + (size_t)d * OUT2DIM + c0;
    atomicAdd(o + 0, v.x * coef);
    atomicAdd(o + 1, v.y * coef);
}

// bias + ELU + f16 requantize (K-swizzled) for layer-2 GEMM input
__global__ void elu_bias_k(const float* __restrict__ agg, const float* __restrict__ b1,
                           _Float16* __restrict__ h2in, long long n)
{
    long long i = (long long)blockIdx.x * blockDim.x + threadIdx.x;
    if (i >= n) return;
    int c = (int)(i & (HID1 - 1));
    float v = agg[i] + b1[c];
    v = v > 0.f ? v : (__expf(v) - 1.f);            // ELU(alpha=1)
    h2in[(i & ~(long long)127) + kswz(c)] = (_Float16)v;
}

// final: sigmoid((out2 + b2) @ lin_w + lin_b)
__global__ void final_k(const float* __restrict__ out2, const float* __restrict__ b2,
                        const float* __restrict__ lw, const float* __restrict__ lb,
                        float* __restrict__ y, int N)
{
    int n = blockIdx.x * blockDim.x + threadIdx.x;
    if (n >= N) return;
    const float* op = out2 + (size_t)n * OUT2DIM;
    float acc = 0.f;
#pragma unroll
    for (int c = 0; c < OUT2DIM; c++) acc += (op[c] + b2[c]) * lw[c];
    acc += lb[0];
    y[n] = 1.f / (1.f + __expf(-acc));
}

// ---------------- launcher ----------------
extern "C" void kernel_launch(void* const* d_in, const int* in_sizes, int n_in,
                              void* d_out, int out_size, void* d_ws, size_t ws_size,
                              hipStream_t stream)
{
    (void)in_sizes; (void)n_in; (void)out_size; (void)ws_size;
    const float*     x     = (const float*)d_in[0];
    const long long* ei    = (const long long*)d_in[1];   // int64 [2,E] flat
    /* d_in[2] edge_attr ignored (edge_dim=None) */
    const float*     W1    = (const float*)d_in[3];
    const float*     asrc1 = (const float*)d_in[4];
    const float*     adst1 = (const float*)d_in[5];
    const float*     b1    = (const float*)d_in[6];
    const float*     W2    = (const float*)d_in[7];
    const float*     asrc2 = (const float*)d_in[8];
    const float*     adst2 = (const float*)d_in[9];
    const float*     b2    = (const float*)d_in[10];
    const float*     lw    = (const float*)d_in[11];
    const float*     lb    = (const float*)d_in[12];
    float* y = (float*)d_out;

    // workspace layout (256B aligned slabs, aliased across phases; peak ~162 MB)
    char* ws = (char*)d_ws;
    size_t off = 0;
    auto take = [&](size_t bytes) -> char* {
        char* p = ws + off; off += (bytes + 255) & ~(size_t)255; return p;
    };
    _Float16* XH   = (_Float16*)take((size_t)NN_NODES * 128 * 2); // also H2IN (layer-2 f16 input)
    _Float16* W1H  = (_Float16*)take(128 * 128 * 2);
    _Float16* W2H  = (_Float16*)take(128 * 64 * 2);
    float*    H1   = (float*)take((size_t)NN_NODES * 128 * 4);    // also G2 (layer-2 projection)
    float*    ES   = (float*)take((size_t)NN_NODES * 4 * 4);      // reused layer2 (N <= N*4)
    float*    ED   = (float*)take((size_t)NN_NODES * 4 * 4);
    float*    AMAX = (float*)take((size_t)NN_NODES * 4 * 4);
    float*    DEN  = (float*)take((size_t)NN_NODES * 4 * 4);
    float*    EX   = (float*)take((size_t)ET_EDGES * 4 * 4);      // reused layer2
    float*    AGG  = (float*)take((size_t)NN_NODES * 128 * 4);    // OUT1, then OUT2
    _Float16* H2IN = XH;
    float*    G2   = H1;

    const int  TB = 256;
    const long long NX  = (long long)NN_NODES * 128;
    const long long NH  = (long long)NN_NODES * 4;
    const dim3 gET((unsigned)((ET_EDGES + TB - 1) / TB));
    const dim3 gAGG((unsigned)((ET_EDGES + 7) / 8));
    const dim3 gGEMM((NN_NODES + 127) / 128);

    // ---- precision conversion ----
    cvt_f16_swz_k<<<dim3((unsigned)((NX + TB - 1) / TB)), TB, 0, stream>>>(x, XH, NX);
    cvt_f16_k<<<dim3((128 * 128 + TB - 1) / TB), TB, 0, stream>>>(W1, W1H, 128 * 128);
    cvt_f16_k<<<dim3((128 * 64 + TB - 1) / TB), TB, 0, stream>>>(W2, W2H, 128 * 64);

    // ---- layer 1 ----
    fill_f32_k<<<dim3((unsigned)((NH + TB - 1) / TB)), TB, 0, stream>>>(AMAX, -3.0e38f, NH);
    fill_f32_k<<<dim3((unsigned)((NH + TB - 1) / TB)), TB, 0, stream>>>(DEN, 0.f, NH);
    fill_f32_k<<<dim3((unsigned)((NX + TB - 1) / TB)), TB, 0, stream>>>(AGG, 0.f, NX);

    gat_gemm_wmma_k<128><<<gGEMM, 256, 0, stream>>>(XH, W1H, H1, NN_NODES);
    nalpha1_k<<<dim3((unsigned)((NH + TB - 1) / TB)), TB, 0, stream>>>(H1, asrc1, adst1, ES, ED, NN_NODES);
    edge_max_k<4><<<gET, TB, 0, stream>>>(ei, NE_EDGES, ET_EDGES, ES, ED, EX, AMAX);
    edge_exp_k<4><<<gET, TB, 0, stream>>>(ei, NE_EDGES, ET_EDGES, EX, AMAX, DEN);
    agg1_k<<<gAGG, TB, 0, stream>>>(ei, NE_EDGES, ET_EDGES, H1, EX, DEN, AGG);
    elu_bias_k<<<dim3((unsigned)((NX + TB - 1) / TB)), TB, 0, stream>>>(AGG, b1, H2IN, NX);

    // ---- layer 2 (H=1, C=64) ----
    const long long N1 = NN_NODES;
    const long long N64 = (long long)NN_NODES * 64;
    fill_f32_k<<<dim3((unsigned)((N1 + TB - 1) / TB)), TB, 0, stream>>>(AMAX, -3.0e38f, N1);
    fill_f32_k<<<dim3((unsigned)((N1 + TB - 1) / TB)), TB, 0, stream>>>(DEN, 0.f, N1);
    fill_f32_k<<<dim3((unsigned)((N64 + TB - 1) / TB)), TB, 0, stream>>>(AGG, 0.f, N64);

    gat_gemm_wmma_k<64><<<gGEMM, 256, 0, stream>>>(H2IN, W2H, G2, NN_NODES);
    nalpha2_k<<<dim3((unsigned)((N1 + TB - 1) / TB)), TB, 0, stream>>>(G2, asrc2, adst2, ES, ED, NN_NODES);
    edge_max_k<1><<<gET, TB, 0, stream>>>(ei, NE_EDGES, ET_EDGES, ES, ED, EX, AMAX);
    edge_exp_k<1><<<gET, TB, 0, stream>>>(ei, NE_EDGES, ET_EDGES, EX, AMAX, DEN);
    agg2_k<<<gAGG, TB, 0, stream>>>(ei, NE_EDGES, ET_EDGES, G2, EX, DEN, AGG);

    // ---- head ----
    final_k<<<dim3((unsigned)((N1 + TB - 1) / TB)), TB, 0, stream>>>(AGG, b2, lw, lb, y, NN_NODES);
}
